// MultiHeadedAttention_65584150609954
// MI455X (gfx1250) — matrix operand order repaired and checked by reading, not compiled, not run on hardware
//
#include <hip/hip_runtime.h>
#include <hip/hip_bf16.h>
#include <math.h>

// ---------------- problem constants ----------------
#define BB   4
#define SS   2048
#define DD   1024
#define HH   16
#define DH   64
#define BS   (BB*SS)          // 8192 rows

typedef _Float16 half8_t  __attribute__((ext_vector_type(8)));
typedef _Float16 v16h     __attribute__((ext_vector_type(16)));
typedef float    v8f      __attribute__((ext_vector_type(8)));
typedef float    float4_t __attribute__((ext_vector_type(4)));

// ---------------- WMMA helpers ----------------
__device__ __forceinline__ v8f wmma_f16(v16h a, v16h b, v8f c) {
  // D = A(16x32 f16) * B(32x16 f16) + C(16x16 f32)
  return __builtin_amdgcn_wmma_f32_16x16x32_f16(false, a, false, b, (short)0, c,
                                                false, false);
}

__device__ __forceinline__ v16h mk16(half8_t lo, half8_t hi) {
  v16h r;
#pragma unroll
  for (int i = 0; i < 8; ++i) { r[i] = lo[i]; r[8 + i] = hi[i]; }
  return r;
}

// A fragment, f16 source, row-major [M][K] with leading dim ld.
// lane&15 = M row; lanes 16-31 take K base +8; element e in {0..7} -> K=kb+ko+e,
// e in {8..15} -> K=kb+16+ko+(e-8).   (ISA 7.12.2, 16-bit A 16x32)
__device__ __forceinline__ v16h load_a_f16(const _Float16* base, int ld, int kb, int lane) {
  int m  = lane & 15;
  int ko = (lane >> 4) * 8;
  const _Float16* p = base + m * ld + kb + ko;
  half8_t lo = *(const half8_t*)(p);
  half8_t hi = *(const half8_t*)(p + 16);
  return mk16(lo, hi);
}

// A fragment from an f32 source (global or LDS), convert on load.
__device__ __forceinline__ v16h load_a_f32(const float* base, int ld, int kb, int lane) {
  int m  = lane & 15;
  int ko = (lane >> 4) * 8;
  const float* p = base + m * ld + kb + ko;
  float4_t x0 = *(const float4_t*)(p);
  float4_t x1 = *(const float4_t*)(p + 4);
  float4_t y0 = *(const float4_t*)(p + 16);
  float4_t y1 = *(const float4_t*)(p + 20);
  v16h r;
#pragma unroll
  for (int i = 0; i < 4; ++i) {
    r[i]      = (_Float16)x0[i];
    r[4 + i]  = (_Float16)x1[i];
    r[8 + i]  = (_Float16)y0[i];
    r[12 + i] = (_Float16)y1[i];
  }
  return r;
}

// B fragment, f16 source stored K-contiguous per column: src[(nb+n)*ld + k].
// lane&15 = column N; lanes 16-31 take K base +8; element e in {0..7} -> K=kb+ko+e,
// e in {8..15} -> K=kb+16+ko+(e-8).
__device__ __forceinline__ v16h load_b_f16(const _Float16* base, int ld, int nb, int kb, int lane) {
  int n  = lane & 15;
  int ko = (lane >> 4) * 8;
  const _Float16* p = base + (size_t)(nb + n) * ld + kb + ko;
  half8_t lo = *(const half8_t*)(p);
  half8_t hi = *(const half8_t*)(p + 16);
  return mk16(lo, hi);
}

// ---------------- kernel 1: weight transpose + f32->f16 ----------------
// out[n*D + k] = (f16) in[k*D + n];  blockIdx.y selects which of 4 matrices
__global__ void wt_transpose_kernel(const float* __restrict__ Wq, const float* __restrict__ Wk,
                                    const float* __restrict__ Wv, const float* __restrict__ Wo,
                                    _Float16* __restrict__ Wqt, _Float16* __restrict__ Wkt,
                                    _Float16* __restrict__ Wvt, _Float16* __restrict__ Wot) {
  int which = blockIdx.y;
  const float* in = (which == 0) ? Wq : (which == 1) ? Wk : (which == 2) ? Wv : Wo;
  _Float16* out   = (which == 0) ? Wqt : (which == 1) ? Wkt : (which == 2) ? Wvt : Wot;
  int idx = blockIdx.x * 256 + threadIdx.x;     // 0 .. D*D-1
  int n = idx >> 10;
  int k = idx & (DD - 1);
  out[idx] = (_Float16)in[k * DD + n];
}

// ---------------- kernel 2: fused QKV projection ----------------
// X[8192x1024] f32  *  Wt(f16, [n][k])  + bias -> f16
//   which 0: Qh  [b,h,s,dh] row-major
//   which 1: Kh  [b,h,s,dh] row-major
//   which 2: VhT [b,h,dh,s] (transposed for PV B-fragments)
// grid (64, 16, 3), block 256 (8 waves); wave owns a 16x64 tile.
__global__ void qkv_proj_kernel(const float* __restrict__ Xq, const float* __restrict__ Xk,
                                const float* __restrict__ Xv,
                                const _Float16* __restrict__ Wqt, const _Float16* __restrict__ Wkt,
                                const _Float16* __restrict__ Wvt,
                                const float* __restrict__ bq, const float* __restrict__ bk,
                                const float* __restrict__ bv,
                                _Float16* __restrict__ Qh, _Float16* __restrict__ Kh,
                                _Float16* __restrict__ VhT) {
  int lane  = threadIdx.x & 31;
  int wave  = threadIdx.x >> 5;
  int mtile = blockIdx.x * 8 + wave;   // 0..511
  int n0    = blockIdx.y * 64;         // 0..960
  int which = blockIdx.z;

  const float*    X    = (which == 0) ? Xq : (which == 1) ? Xk : Xv;
  const _Float16* Wt   = (which == 0) ? Wqt : (which == 1) ? Wkt : Wvt;
  const float*    bias = (which == 0) ? bq : (which == 1) ? bk : bv;

  const float* Arow = X + (size_t)mtile * 16 * DD;

  v8f acc0 = {}, acc1 = {}, acc2 = {}, acc3 = {};
  for (int kc = 0; kc < DD; kc += 32) {
    v16h a  = load_a_f32(Arow, DD, kc, lane);
    acc0 = wmma_f16(a, load_b_f16(Wt, DD, n0 +  0, kc, lane), acc0);
    acc1 = wmma_f16(a, load_b_f16(Wt, DD, n0 + 16, kc, lane), acc1);
    acc2 = wmma_f16(a, load_b_f16(Wt, DD, n0 + 32, kc, lane), acc2);
    acc3 = wmma_f16(a, load_b_f16(Wt, DD, n0 + 48, kc, lane), acc3);
  }

  v8f accs[4] = {acc0, acc1, acc2, acc3};
  int nl = lane & 15;
  int mo = (lane >> 4) * 8;
#pragma unroll
  for (int sub = 0; sub < 4; ++sub) {
    int nabs = n0 + sub * 16 + nl;
    float bval = bias[nabs];
    int h = nabs >> 6, d = nabs & 63;
#pragma unroll
    for (int i = 0; i < 8; ++i) {
      int r = mtile * 16 + mo + i;          // global row = b*S + s
      int b = r >> 11, s = r & (SS - 1);
      _Float16 val = (_Float16)(accs[sub][i] + bval);
      if (which == 2)
        VhT[((size_t)(b * HH + h) * DH + d) * SS + s] = val;
      else if (which == 0)
        Qh[((size_t)(b * HH + h) * SS + s) * DH + d] = val;
      else
        Kh[((size_t)(b * HH + h) * SS + s) * DH + d] = val;
    }
  }
}

// ---------------- kernel 3: fused scores + softmax + PV ----------------
// One workgroup per (qtile of 16 queries, bh). The full 16x2048 score strip
// lives in LDS (128 KB of the WGP's 320 KB):
//   phase 1: 8 waves x WMMA -> masked scores in LDS
//   phase 2: row softmax in LDS (shfl reductions, 16 lanes per row)
//   phase 3: single non-temporal write of probs to d_out (only HBM touch)
//   phase 4: PV with A-fragments read from LDS, B from VhT (L2-resident)
//   phase 5: cross-wave partial reduction -> Hc (f16)
// grid (128, 64), block 256, dynamic LDS = (16*2048 + 8*256)*4 bytes.
__global__ void attn_fused_kernel(const _Float16* __restrict__ Qh, const _Float16* __restrict__ Kh,
                                  const _Float16* __restrict__ VhT,
                                  const unsigned char* __restrict__ mask,
                                  const int* __restrict__ reverse,
                                  float* __restrict__ probs, _Float16* __restrict__ Hc) {
  extern __shared__ float lds[];
  float* Ssc  = lds;              // [16][2048] scores -> probs
  float* Pred = lds + 16 * SS;    // [8][16][16] PV partials

  int t    = threadIdx.x;
  int lane = t & 31;
  int wave = t >> 5;
  int qtile = blockIdx.x;
  int bh    = blockIdx.y;
  int q0 = qtile * 16;
  int b  = bh >> 4, h = bh & 15;
  int rev = reverse[0];
  int nl = lane & 15;
  int mo = (lane >> 4) * 8;

  // hoist the Q A-fragments: reused for all 128 K-tiles
  const _Float16* Qb = Qh + ((size_t)bh * SS + q0) * DH;
  v16h aq0 = load_a_f16(Qb, DH, 0,  lane);
  v16h aq1 = load_a_f16(Qb, DH, 32, lane);

  // ---- phase 1: masked scores into LDS ----
  for (int kt = wave; kt < SS / 16; kt += 8) {
    int k0 = kt * 16;
    bool fully = rev ? (k0 + 15 < q0) : (k0 > q0 + 15);   // uniform per wave
    if (fully) {
#pragma unroll
      for (int i = 0; i < 8; ++i) Ssc[(mo + i) * SS + k0 + nl] = -1.0e9f;
      continue;
    }
    const _Float16* Kb = Kh + ((size_t)bh * SS + k0) * DH;
    v8f acc = {};
    acc = wmma_f16(aq0, load_b_f16(Kb, DH, 0, 0,  lane), acc);
    acc = wmma_f16(aq1, load_b_f16(Kb, DH, 0, 32, lane), acc);
#pragma unroll
    for (int i = 0; i < 8; ++i) {
      int qi = q0 + mo + i;
      int ki = k0 + nl;
      bool msk = (mask[b * SS + qi] != 0) || (rev ? (ki < qi) : (ki > qi));
      Ssc[(mo + i) * SS + k0 + nl] = msk ? -1.0e9f : acc[i] * 0.125f;
    }
  }
  __syncthreads();

  // ---- phase 2: softmax per row, fully in LDS ----
  {
    int row = t >> 4;          // 0..15
    int sub = t & 15;          // 16 lanes cooperate on one row
    float* rp = Ssc + row * SS;
    float mx = -INFINITY;
    for (int j = 0; j < SS / 16; ++j) mx = fmaxf(mx, rp[sub + j * 16]);
#pragma unroll
    for (int off = 8; off > 0; off >>= 1) mx = fmaxf(mx, __shfl_xor(mx, off, 16));
    float sum = 0.f;
    for (int j = 0; j < SS / 16; ++j) {
      float e = __expf(rp[sub + j * 16] - mx);
      rp[sub + j * 16] = e;
      sum += e;
    }
#pragma unroll
    for (int off = 8; off > 0; off >>= 1) sum += __shfl_xor(sum, off, 16);
    float inv = 1.0f / sum;
    for (int j = 0; j < SS / 16; ++j) rp[sub + j * 16] *= inv;
  }
  __syncthreads();

  // ---- phase 3: single streaming write of probs (non-temporal, bypass L2 reuse) ----
  {
    float* pg = probs + ((size_t)bh * SS + q0) * SS;   // rows contiguous: flat copy
    for (int idx = t; idx < 16 * SS; idx += 256)
      __builtin_nontemporal_store(Ssc[idx], pg + idx);
  }

  // ---- phase 4: PV, A-fragments from LDS probs ----
  {
    int sub  = wave & 3;   // dh subtile (16 cols)
    int half = wave >> 2;  // key-range half
    const _Float16* Vb = VhT + (size_t)bh * DH * SS;
    v8f acc = {};
    int kend = (half + 1) * (SS / 2);
    for (int kc = half * (SS / 2); kc < kend; kc += 32) {
      v16h a = load_a_f32(Ssc, SS, kc, lane);   // LDS read + cvt
      acc = wmma_f16(a, load_b_f16(Vb, SS, sub * 16, kc, lane), acc);
    }
#pragma unroll
    for (int i = 0; i < 8; ++i)
      Pred[wave * 256 + (mo + i) * 16 + nl] = acc[i];
  }
  __syncthreads();

  // ---- phase 5: reduce key-halves, store Hc (f16) ----
  for (int idx = t; idx < 16 * DH; idx += 256) {
    int m = idx >> 6, n = idx & 63;
    int sub = n >> 4, n16 = n & 15;
    float sum = Pred[sub * 256 + m * 16 + n16] + Pred[(4 + sub) * 256 + m * 16 + n16];
    Hc[(size_t)(b * SS + q0 + m) * DD + h * 64 + n] = (_Float16)sum;
  }
}

// ---------------- kernel 4: output projection -> f32 hidden ----------------
// grid (64, 16), block 256; wave owns 16x64 tile.
__global__ void out_proj_kernel(const _Float16* __restrict__ Hc, const _Float16* __restrict__ Wot,
                                const float* __restrict__ bo, float* __restrict__ out) {
  int lane  = threadIdx.x & 31;
  int wave  = threadIdx.x >> 5;
  int mtile = blockIdx.x * 8 + wave;
  int n0    = blockIdx.y * 64;

  const _Float16* Arow = Hc + (size_t)mtile * 16 * DD;

  v8f acc0 = {}, acc1 = {}, acc2 = {}, acc3 = {};
  for (int kc = 0; kc < DD; kc += 32) {
    v16h a = load_a_f16(Arow, DD, kc, lane);
    acc0 = wmma_f16(a, load_b_f16(Wot, DD, n0 +  0, kc, lane), acc0);
    acc1 = wmma_f16(a, load_b_f16(Wot, DD, n0 + 16, kc, lane), acc1);
    acc2 = wmma_f16(a, load_b_f16(Wot, DD, n0 + 32, kc, lane), acc2);
    acc3 = wmma_f16(a, load_b_f16(Wot, DD, n0 + 48, kc, lane), acc3);
  }

  v8f accs[4] = {acc0, acc1, acc2, acc3};
  int nl = lane & 15;
  int mo = (lane >> 4) * 8;
#pragma unroll
  for (int sub = 0; sub < 4; ++sub) {
    int nabs = n0 + sub * 16 + nl;
    float bval = bo[nabs];
#pragma unroll
    for (int i = 0; i < 8; ++i) {
      int r = mtile * 16 + mo + i;
      out[(size_t)r * DD + nabs] = accs[sub][i] + bval;
    }
  }
}

// ---------------- host launch ----------------
extern "C" void kernel_launch(void* const* d_in, const int* in_sizes, int n_in,
                              void* d_out, int out_size, void* d_ws, size_t ws_size,
                              hipStream_t stream) {
  // inputs (setup_inputs order)
  const float*         q    = (const float*)d_in[0];
  const float*         k    = (const float*)d_in[1];
  const float*         v    = (const float*)d_in[2];
  const unsigned char* mask = (const unsigned char*)d_in[3];   // jax bool -> 1 byte
  const float* Wq = (const float*)d_in[4];
  const float* bq = (const float*)d_in[5];
  const float* Wk = (const float*)d_in[6];
  const float* bk = (const float*)d_in[7];
  const float* Wv = (const float*)d_in[8];
  const float* bv = (const float*)d_in[9];
  const float* Wo = (const float*)d_in[10];
  const float* bo = (const float*)d_in[11];
  const int*   reverse = (const int*)d_in[12];

  // outputs: hidden [B,S,D] f32, then prob_attn [B,H,S,S] f32
  float* hidden = (float*)d_out;
  float* probs  = (float*)d_out + (size_t)BB * SS * DD;

  // workspace layout (f16 elements)
  _Float16* ws  = (_Float16*)d_ws;
  const size_t WSZ = (size_t)DD * DD;           // 1,048,576
  const size_t TSZ = (size_t)BB * HH * SS * DH; // 8,388,608
  _Float16* Wqt = ws;
  _Float16* Wkt = Wqt + WSZ;
  _Float16* Wvt = Wkt + WSZ;
  _Float16* Wot = Wvt + WSZ;
  _Float16* Qh  = Wot + WSZ;
  _Float16* Kh  = Qh + TSZ;
  _Float16* VhT = Kh + TSZ;
  _Float16* Hc  = VhT + TSZ;

  const int ATTN_LDS = (16 * SS + 8 * 256) * sizeof(float);   // 139,264 B
  // allow >64KB dynamic LDS (320KB/WGP on CDNA5)
  (void)hipFuncSetAttribute(reinterpret_cast<const void*>(attn_fused_kernel),
                            hipFuncAttributeMaxDynamicSharedMemorySize, ATTN_LDS);

  // 1) weight transpose + convert
  wt_transpose_kernel<<<dim3((DD * DD) / 256, 4), 256, 0, stream>>>(
      Wq, Wk, Wv, Wo, Wqt, Wkt, Wvt, Wot);

  // 2) fused QKV projections
  qkv_proj_kernel<<<dim3(BS / 16 / 8, DD / 64, 3), 256, 0, stream>>>(
      q, k, v, Wqt, Wkt, Wvt, bq, bk, bv, Qh, Kh, VhT);

  // 3) fused scores + softmax + PV (probs written to HBM exactly once)
  attn_fused_kernel<<<dim3(SS / 16, BB * HH), 256, ATTN_LDS, stream>>>(
      Qh, Kh, VhT, mask, reverse, probs, Hc);

  // 4) output projection
  out_proj_kernel<<<dim3(BS / 16 / 8, DD / 64), 256, 0, stream>>>(Hc, Wot, bo, hidden);

  (void)in_sizes; (void)n_in; (void)out_size; (void)ws_size;
}
